// V1NetCell_36369783062529
// MI455X (gfx1250) — matrix operand order repaired and checked by reading, not compiled, not run on hardware
//
#include <hip/hip_runtime.h>
#include <cstdint>
#include <cstddef>

// ---------------------------------------------------------------------------
// V1NetCell fused kernel for gfx1250 (MI455X): depthwise convs in VALU,
// pointwise convs via v_wmma_f32_16x16x32_bf16, fused gates + c_next,
// deterministic 2-stage LayerNorm reduction, vectorized h_next pass.
// ---------------------------------------------------------------------------

typedef __attribute__((ext_vector_type(16))) __bf16 v16bf;
typedef __attribute__((ext_vector_type(8)))  float  v8f;

#define B_DIM 8
#define C_DIM 128
#define HW    128
#define N_PER_SAMPLE (C_DIM * HW * HW)        // 2,097,152  (== 1<<21)
#define N_TOTAL      (B_DIM * N_PER_SAMPLE)   // 16,777,216

// LDS geometry (bf16 elements). Channel-dim padded so that consecutive pixels
// rotate by 4 banks -> conflict-free ds_load_b128 across 16 lanes.
#define XH_STRIDE 264   // 256 + 8 pad ; row = 528B (16B aligned)
#define H_STRIDE  136   // 128 + 8 pad ; row = 272B (16B aligned)
#define LDS_XH_OFF 0
#define LDS_EX_OFF (128 * XH_STRIDE * 2)                 // 67584
#define LDS_IN_OFF (LDS_EX_OFF + 128 * H_STRIDE * 2)     // 102400
#define LDS_RED_OFF (LDS_IN_OFF + 128 * H_STRIDE * 2)    // 137216
#define LDS_BYTES  (LDS_RED_OFF + 512 * 4)               // 139264 (< 160KB)

__device__ __forceinline__ unsigned short f2bf(float f) {
  unsigned u = __float_as_uint(f);
  u += 0x7FFFu + ((u >> 16) & 1u);          // round-to-nearest-even
  return (unsigned short)(u >> 16);
}

__device__ __forceinline__ float sigm(float x) {
  return 1.0f / (1.0f + __expf(-x));
}

union FragU { uint4 q[2]; v16bf v; };

__device__ __forceinline__ v16bf load_frag16(const unsigned short* p) {
  FragU u;
  const uint4* q = (const uint4*)p;   // 16B-aligned by construction
  u.q[0] = q[0];
  u.q[1] = q[1];
  return u.v;
}

// ---------------------------------------------------------------------------
// Weight packing: f32 [O x K] row-major  ->  bf16 WMMA A-fragments.
// Fragment (otile,ktile) holds 512 bf16: lane L owns row m=L%16;
//   lane<16 : elements e -> K = e       (K 0..15)
//   lane>=16: elements e -> K = e+16    (K 16..31)
// (contiguous K order, matching the B-fragment LDS loads.)
// ---------------------------------------------------------------------------
#define PACK_XH  (384 * 256)   // 98304
#define PACK_H   (128 * 128)   // 16384
#define PACK_TOTAL (PACK_XH + 2 * PACK_H)

__global__ void v1net_pack(const float* __restrict__ wxh,
                           const float* __restrict__ wexc,
                           const float* __restrict__ winh,
                           unsigned short* __restrict__ axh,
                           unsigned short* __restrict__ aexc,
                           unsigned short* __restrict__ ainh) {
  int i = blockIdx.x * 256 + threadIdx.x;
  if (i >= PACK_TOTAL) return;
  const float* W;
  unsigned short* dst;
  int Kdim, nkt, d;
  if (i < PACK_XH)            { W = wxh;  dst = axh;  Kdim = 256; nkt = 8; d = i; }
  else if (i < PACK_XH+PACK_H){ W = wexc; dst = aexc; Kdim = 128; nkt = 4; d = i - PACK_XH; }
  else                        { W = winh; dst = ainh; Kdim = 128; nkt = 4; d = i - PACK_XH - PACK_H; }
  int frag = d >> 9, within = d & 511;
  int lane = within >> 4, e = within & 15;
  int otile = frag / nkt, ktile = frag % nkt;
  int m = lane & 15;
  int kin = (lane < 16) ? e : e + 16;
  int o = otile * 16 + m;
  int k = ktile * 32 + kin;
  dst[d] = f2bf(W[o * Kdim + k]);
}

// ---------------------------------------------------------------------------
// Depthwise KxK conv on one image row, one channel per thread, sliding-window
// registers. Pixel loop is blocked by K with the inner K iterations fully
// unrolled so the window rotation becomes static SSA renaming (no v_mov
// shuffle per pixel). Result -> LDS as bf16 [px][ch].
// ---------------------------------------------------------------------------
template<int K>
__device__ __forceinline__ void dwconv_row(const float* __restrict__ plane, int y,
                                           const float* __restrict__ wsrc, float bias,
                                           unsigned short* dst, int stride, int cch) {
  constexpr int HALF = K / 2;
  float wgt[K * K];
#pragma unroll
  for (int i = 0; i < K * K; ++i) wgt[i] = wsrc[i];

  const float* rowp[K];
  bool ok[K];
#pragma unroll
  for (int r = 0; r < K; ++r) {
    int yy = y + r - HALF;
    ok[r] = (yy >= 0) && (yy < HW);
    rowp[r] = plane + (ok[r] ? yy : 0) * HW;
  }

  float win[K][K];
#pragma unroll
  for (int r = 0; r < K; ++r)
#pragma unroll
    for (int j = 0; j < K - 1; ++j) {
      int xx = j - HALF;                       // columns for px=0 (leftmost K-1)
      win[r][j] = (ok[r] && xx >= 0) ? rowp[r][xx] : 0.0f;
    }

  auto body = [&](int px) {
    int xx = px + HALF;
    bool xok = (xx < HW);
#pragma unroll
    for (int r = 0; r < K; ++r)
      win[r][K - 1] = (ok[r] && xok) ? rowp[r][xx] : 0.0f;

    float acc = bias;
#pragma unroll
    for (int r = 0; r < K; ++r)
#pragma unroll
      for (int j = 0; j < K; ++j)
        acc = fmaf(win[r][j], wgt[r * K + j], acc);

    dst[px * stride + cch] = f2bf(acc);

#pragma unroll
    for (int r = 0; r < K; ++r)
#pragma unroll
      for (int j = 0; j < K - 1; ++j)
        win[r][j] = win[r][j + 1];
  };

  int px = 0;
  for (int blk = 0; blk < HW / K; ++blk) {     // K-blocked: rotation is free
#pragma unroll
    for (int j = 0; j < K; ++j) {
      body(px + j);
    }
    px += K;
  }
  for (; px < HW; ++px) body(px);              // tail (HW % K pixels)
}

// ---------------------------------------------------------------------------
// Main fused kernel: 1 block = 1 (batch, row). 256 threads = 8 wave32.
// ---------------------------------------------------------------------------
__launch_bounds__(256)
__global__ void v1net_main(const float* __restrict__ x,
                           const float* __restrict__ h,
                           const float* __restrict__ c,
                           const float* __restrict__ w_xh_dw, const float* __restrict__ b_xh_dw,
                           const float* __restrict__ b_xh_pw,
                           const float* __restrict__ w_exc_dw, const float* __restrict__ b_exc_dw,
                           const float* __restrict__ b_exc_pw,
                           const float* __restrict__ w_inh_dw, const float* __restrict__ b_inh_dw,
                           const float* __restrict__ b_inh_pw,
                           const unsigned short* __restrict__ axh,
                           const unsigned short* __restrict__ aexc,
                           const unsigned short* __restrict__ ainh,
                           float* __restrict__ out,      // [0,N): o-gate stash, [N,2N): c_next
                           float* __restrict__ part) {   // per-block (sum, sumsq)
  extern __shared__ char smem[];
  unsigned short* lds_xh = (unsigned short*)(smem + LDS_XH_OFF);
  unsigned short* lds_ex = (unsigned short*)(smem + LDS_EX_OFF);
  unsigned short* lds_in = (unsigned short*)(smem + LDS_IN_OFF);
  float* red = (float*)(smem + LDS_RED_OFF);

  const int bid = blockIdx.x;
  const int b = bid >> 7;          // batch
  const int y = bid & 127;         // row
  const int tid = threadIdx.x;

  // ---- Phase 1: depthwise convs -> LDS (bf16) --------------------------
  {   // concat(x,h) branch, 5x5: channel = tid (0..255)
    int cc = tid;
    const float* plane = (cc < C_DIM)
        ? x + (size_t)(b * C_DIM + cc) * (HW * HW)
        : h + (size_t)(b * C_DIM + (cc - C_DIM)) * (HW * HW);
    dwconv_row<5>(plane, y, w_xh_dw + cc * 25, b_xh_dw[cc], lds_xh, XH_STRIDE, cc);
  }
  if (tid < 128) {  // waves 0..3: excitation 7x7 on h
    int cc = tid;
    const float* plane = h + (size_t)(b * C_DIM + cc) * (HW * HW);
    dwconv_row<7>(plane, y, w_exc_dw + cc * 49, b_exc_dw[cc], lds_ex, H_STRIDE, cc);
  } else {          // waves 4..7: inhibition 5x5 on h
    int cc = tid - 128;
    const float* plane = h + (size_t)(b * C_DIM + cc) * (HW * HW);
    dwconv_row<5>(plane, y, w_inh_dw + cc * 25, b_inh_dw[cc], lds_in, H_STRIDE, cc);
  }
  __syncthreads();

  // ---- Phase 2: pointwise GEMMs via WMMA + gate epilogue ---------------
  const int wv   = tid >> 5;          // wave id = pixel tile (16 px)
  const int lane = tid & 31;
  const bool hi  = lane >= 16;
  const int ln   = lane & 15;
  const int px   = wv * 16 + ln;
  const int bko  = hi ? 16 : 0;       // B-fragment K sub-offset per lane half

  float ssum = 0.0f, ssq = 0.0f;
  float* out_o = out;
  float* out_c = out + (size_t)N_TOTAL;

  for (int jt = 0; jt < 8; ++jt) {    // 8 output-channel tiles of 16
    __builtin_prefetch(axh + ((size_t)jt * 8) * 512, 0, 0);   // global_prefetch_b8

    v8f acc_f = {0,0,0,0,0,0,0,0};
    v8f acc_g = {0,0,0,0,0,0,0,0};
    v8f acc_o = {0,0,0,0,0,0,0,0};
    v8f acc_e = {0,0,0,0,0,0,0,0};
    v8f acc_i = {0,0,0,0,0,0,0,0};

    // xh branch: K = 256, W rows {jt, 8+jt, 16+jt} (f_g, g_g, o_g splits)
#pragma unroll 4
    for (int kt = 0; kt < 8; ++kt) {
      v16bf bf = load_frag16(lds_xh + px * XH_STRIDE + kt * 32 + bko);
      v16bf af = load_frag16(axh + ((size_t)(jt)      * 8 + kt) * 512 + lane * 16);
      v16bf ag = load_frag16(axh + ((size_t)(8 + jt)  * 8 + kt) * 512 + lane * 16);
      v16bf ao = load_frag16(axh + ((size_t)(16 + jt) * 8 + kt) * 512 + lane * 16);
      acc_f = __builtin_amdgcn_wmma_f32_16x16x32_bf16(false, af, false, bf, (short)0, acc_f, false, false);
      acc_g = __builtin_amdgcn_wmma_f32_16x16x32_bf16(false, ag, false, bf, (short)0, acc_g, false, false);
      acc_o = __builtin_amdgcn_wmma_f32_16x16x32_bf16(false, ao, false, bf, (short)0, acc_o, false, false);
    }
    // exc / inh branches: K = 128
#pragma unroll 4
    for (int kt = 0; kt < 4; ++kt) {
      v16bf be = load_frag16(lds_ex + px * H_STRIDE + kt * 32 + bko);
      v16bf ae = load_frag16(aexc + ((size_t)jt * 4 + kt) * 512 + lane * 16);
      acc_e = __builtin_amdgcn_wmma_f32_16x16x32_bf16(false, ae, false, be, (short)0, acc_e, false, false);
      v16bf bi = load_frag16(lds_in + px * H_STRIDE + kt * 32 + bko);
      v16bf ai = load_frag16(ainh + ((size_t)jt * 4 + kt) * 512 + lane * 16);
      acc_i = __builtin_amdgcn_wmma_f32_16x16x32_bf16(false, ai, false, bi, (short)0, acc_i, false, false);
    }

    // Epilogue: D-tile element (r, lane) is (M = r + 8*hi, N = ln).
#pragma unroll
    for (int r = 0; r < 8; ++r) {
      int ch = jt * 16 + r + (hi ? 8 : 0);
      float fg = acc_f[r] + b_xh_pw[ch];
      float gg = acc_g[r] + b_xh_pw[128 + ch];
      float og = acc_o[r] + b_xh_pw[256 + ch];
      float re = acc_e[r] + b_exc_pw[ch];
      float ri = acc_i[r] + b_inh_pw[ch];

      float f = sigm(fg);
      float o = sigm(og);
      float g = sigm(ri) * (fmaxf(gg, 0.0f) + sigm(re));

      size_t idx = ((size_t)(b * C_DIM + ch) * HW + y) * HW + px;
      float cv = c[idx];
      float cn = fmaf(f, cv, (1.0f - f) * g);
      out_c[idx] = cn;
      out_o[idx] = o;                 // stash o-gate in h_next slot
      ssum += cn;
      ssq  = fmaf(cn, cn, ssq);
    }
  }

  // ---- Per-block reduction of (sum, sumsq) for LayerNorm ---------------
  red[tid] = ssum;
  red[256 + tid] = ssq;
  __syncthreads();
  for (int s = 128; s > 0; s >>= 1) {
    if (tid < s) {
      red[tid] += red[tid + s];
      red[256 + tid] += red[256 + tid + s];
    }
    __syncthreads();
  }
  if (tid == 0) {
    part[bid * 2 + 0] = red[0];
    part[bid * 2 + 1] = red[256];
  }
}

// ---------------------------------------------------------------------------
// LayerNorm stats: deterministic fixed-order sum of 128 row-partials/sample.
// ---------------------------------------------------------------------------
__global__ void v1net_stats(const float* __restrict__ part, float* __restrict__ stats) {
  int t = threadIdx.x;
  if (t >= B_DIM) return;
  float s = 0.0f, q = 0.0f;
  for (int i = 0; i < 128; ++i) {
    s += part[(t * 128 + i) * 2 + 0];
    q += part[(t * 128 + i) * 2 + 1];
  }
  const float invN = 1.0f / (float)N_PER_SAMPLE;
  float mean = s * invN;
  float var = q * invN - mean * mean;
  stats[t] = mean;
  stats[B_DIM + t] = rsqrtf(var + 1e-5f);
}

// ---------------------------------------------------------------------------
// h_next = o * relu((c_next - mean) * rsqrt(var+eps)), float4-vectorized,
// in-place over the o-gate stash.
// ---------------------------------------------------------------------------
__global__ void v1net_final(const float* __restrict__ stats, float* __restrict__ out) {
  size_t i4 = ((size_t)blockIdx.x * 256 + threadIdx.x) * 4;
  if (i4 >= (size_t)N_TOTAL) return;
  int b = (int)(i4 >> 21);            // N_PER_SAMPLE == 1<<21, divisible by 4
  float mean = stats[b];
  float inv = stats[B_DIM + b];
  float4 cv = *(const float4*)(out + (size_t)N_TOTAL + i4);
  float4 ov = *(float4*)(out + i4);
  float4 r;
  r.x = ov.x * fmaxf((cv.x - mean) * inv, 0.0f);
  r.y = ov.y * fmaxf((cv.y - mean) * inv, 0.0f);
  r.z = ov.z * fmaxf((cv.z - mean) * inv, 0.0f);
  r.w = ov.w * fmaxf((cv.w - mean) * inv, 0.0f);
  *(float4*)(out + i4) = r;
}

// ---------------------------------------------------------------------------
extern "C" void kernel_launch(void* const* d_in, const int* in_sizes, int n_in,
                              void* d_out, int out_size, void* d_ws, size_t ws_size,
                              hipStream_t stream) {
  const float* x        = (const float*)d_in[0];
  const float* h        = (const float*)d_in[1];
  const float* c        = (const float*)d_in[2];
  const float* w_xh_dw  = (const float*)d_in[3];
  const float* b_xh_dw  = (const float*)d_in[4];
  const float* w_xh_pw  = (const float*)d_in[5];
  const float* b_xh_pw  = (const float*)d_in[6];
  const float* w_exc_dw = (const float*)d_in[7];
  const float* b_exc_dw = (const float*)d_in[8];
  const float* w_exc_pw = (const float*)d_in[9];
  const float* b_exc_pw = (const float*)d_in[10];
  const float* w_inh_dw = (const float*)d_in[11];
  const float* b_inh_dw = (const float*)d_in[12];
  const float* w_inh_pw = (const float*)d_in[13];
  const float* b_inh_pw = (const float*)d_in[14];

  float* out = (float*)d_out;

  char* ws = (char*)d_ws;
  float* part  = (float*)ws;                         //  8 KB : 1024 x {sum,sumsq}
  float* stats = (float*)(ws + 8192);                //  64 B : mean[8], inv[8]
  unsigned short* axh  = (unsigned short*)(ws + 16384);
  unsigned short* aexc = axh + PACK_XH;
  unsigned short* ainh = aexc + PACK_H;              // total ws use ~272 KB

  // 1) pack pointwise weights into bf16 WMMA A-fragment layout
  v1net_pack<<<(PACK_TOTAL + 255) / 256, 256, 0, stream>>>(
      w_xh_pw, w_exc_pw, w_inh_pw, axh, aexc, ainh);

  // 2) fused depthwise + WMMA pointwise + gates + c_next (+ o stash, partials)
  v1net_main<<<B_DIM * HW, 256, LDS_BYTES, stream>>>(
      x, h, c,
      w_xh_dw, b_xh_dw, b_xh_pw,
      w_exc_dw, b_exc_dw, b_exc_pw,
      w_inh_dw, b_inh_dw, b_inh_pw,
      axh, aexc, ainh,
      out, part);

  // 3) per-sample LayerNorm stats (deterministic order)
  v1net_stats<<<1, 32, 0, stream>>>(part, stats);

  // 4) h_next elementwise
  v1net_final<<<N_TOTAL / 4 / 256, 256, 0, stream>>>(stats, out);
}
